// metal_attention_79044578115933
// MI455X (gfx1250) — matrix operand order repaired and checked by reading, not compiled, not run on hardware
//
#include <hip/hip_runtime.h>

typedef __attribute__((ext_vector_type(16))) __bf16 v16bf;
typedef __attribute__((ext_vector_type(8)))  __bf16 v8bf;
typedef __attribute__((ext_vector_type(8)))  float  v8f;

#define DF 512      // d_feature == d_k
#define NKEY 2048
#define BQ 128      // Lq
#define KBLK 128    // keys per flash iteration

__device__ __forceinline__ v16bf combine8(v8bf lo, v8bf hi) {
  return __builtin_shufflevector(lo, hi, 0,1,2,3,4,5,6,7,8,9,10,11,12,13,14,15);
}

__device__ __forceinline__ v8bf cvt8(float4 a, float4 b) {
  v8bf p;
  p[0] = (__bf16)a.x; p[1] = (__bf16)a.y; p[2] = (__bf16)a.z; p[3] = (__bf16)a.w;
  p[4] = (__bf16)b.x; p[5] = (__bf16)b.y; p[6] = (__bf16)b.z; p[7] = (__bf16)b.w;
  return p;
}

__device__ __forceinline__ v8f wmma_bf16(v16bf a, v16bf b, v8f c) {
  return __builtin_amdgcn_wmma_f32_16x16x32_bf16(false, a, false, b, (short)0, c,
                                                 false, false);
}

// ---------------------------------------------------------------------------
// fp32 -> bf16 bulk convert (used for the three weight matrices; 8 elem/thread)
// ---------------------------------------------------------------------------
__global__ __launch_bounds__(256)
void cvt_f32_bf16(const float* __restrict__ src, __bf16* __restrict__ dst, int n8) {
  int i = blockIdx.x * 256 + threadIdx.x;
  if (i < n8) {
    const float4* s = (const float4*)src + 2 * (size_t)i;
    *(v8bf*)(dst + (size_t)i * 8) = cvt8(s[0], s[1]);
  }
}

// ---------------------------------------------------------------------------
// Q projection: Y = X @ W^T  (X fp32 [M,512], W bf16 [512,512], Y bf16 row-major)
// Block = 256 threads (8 waves): 16 M-rows x 512 N-cols; X tile staged in LDS.
// ---------------------------------------------------------------------------
__global__ __launch_bounds__(256)
void q_gemm(const float* __restrict__ X, const __bf16* __restrict__ W,
            __bf16* __restrict__ Y) {
  __shared__ __bf16 Xs[16][DF];                 // 16 KB
  const int tid  = threadIdx.x;
  const int wave = tid >> 5;
  const int lane = tid & 31;
  const int h    = lane >> 4;
  const int ln   = lane & 15;
  const int mb   = blockIdx.x * 16;
  const int nb   = wave * 64;

  {
    const float4* src = (const float4*)(X + (size_t)mb * DF);
    for (int i = tid; i < (16 * DF) / 8; i += 256)
      *(v8bf*)(&Xs[0][0] + i * 8) = cvt8(src[2 * i], src[2 * i + 1]);
  }
  __syncthreads();

  v8f acc[4];
#pragma unroll
  for (int t = 0; t < 4; ++t)
#pragma unroll
    for (int v = 0; v < 8; ++v) acc[t][v] = 0.0f;

  for (int fb = 0; fb < DF; fb += 32) {
    v8bf r0 = *(const v8bf*)(&Xs[ln][fb + 16 * h]);
    v8bf r1 = *(const v8bf*)(&Xs[ln][fb + 16 * h + 8]);
    v16bf a = combine8(r0, r1);
#pragma unroll
    for (int nt = 0; nt < 4; ++nt) {
      v16bf b = *(const v16bf*)(W + (size_t)(nb + nt * 16 + ln) * DF + fb + 16 * h);
      acc[nt] = wmma_bf16(a, b, acc[nt]);
    }
  }
#pragma unroll
  for (int nt = 0; nt < 4; ++nt) {
    int col = nb + nt * 16 + ln;
#pragma unroll
    for (int v = 0; v < 8; ++v)            // C/D: M = v + 8h, N = lane%16
      Y[(size_t)(mb + 8 * h + v) * DF + col] = (__bf16)acc[nt][v];
  }
}

// ---------------------------------------------------------------------------
// Fused K + V projection: K row-major bf16, V stored transposed per batch
// Vt[b][d][key].  One X-tile stage feeds 8 WMMAs per K-step per wave; x is
// read from HBM exactly once.
// ---------------------------------------------------------------------------
__global__ __launch_bounds__(256)
void kv_gemm(const float* __restrict__ X, const __bf16* __restrict__ Wk,
             const __bf16* __restrict__ Wv, __bf16* __restrict__ K,
             __bf16* __restrict__ Vt) {
  __shared__ __bf16 Xs[16][DF];                 // 16 KB
  const int tid  = threadIdx.x;
  const int wave = tid >> 5;
  const int lane = tid & 31;
  const int h    = lane >> 4;
  const int ln   = lane & 15;
  const int mb   = blockIdx.x * 16;
  const int nb   = wave * 64;

  {
    const float4* src = (const float4*)(X + (size_t)mb * DF);
    for (int i = tid; i < (16 * DF) / 8; i += 256)
      *(v8bf*)(&Xs[0][0] + i * 8) = cvt8(src[2 * i], src[2 * i + 1]);
  }
  __syncthreads();

  v8f kacc[4], vacc[4];
#pragma unroll
  for (int t = 0; t < 4; ++t)
#pragma unroll
    for (int v = 0; v < 8; ++v) { kacc[t][v] = 0.0f; vacc[t][v] = 0.0f; }

  for (int fb = 0; fb < DF; fb += 32) {
    v8bf r0 = *(const v8bf*)(&Xs[ln][fb + 16 * h]);
    v8bf r1 = *(const v8bf*)(&Xs[ln][fb + 16 * h + 8]);
    v16bf a = combine8(r0, r1);
#pragma unroll
    for (int nt = 0; nt < 4; ++nt) {
      size_t woff = (size_t)(nb + nt * 16 + ln) * DF + fb + 16 * h;
      v16bf bk = *(const v16bf*)(Wk + woff);
      kacc[nt] = wmma_bf16(a, bk, kacc[nt]);
      v16bf bv = *(const v16bf*)(Wv + woff);
      vacc[nt] = wmma_bf16(a, bv, vacc[nt]);
    }
  }

  // K: row-major store
#pragma unroll
  for (int nt = 0; nt < 4; ++nt) {
    int col = nb + nt * 16 + ln;
#pragma unroll
    for (int v = 0; v < 8; ++v)
      K[(size_t)(mb + 8 * h + v) * DF + col] = (__bf16)kacc[nt][v];
  }
  // Vt[b][d][key]: lane's 8 consecutive M values -> one 16B store per tile
  {
    int row0  = mb + 8 * h;
    int batch = row0 / NKEY;
    int key0  = row0 % NKEY;
#pragma unroll
    for (int nt = 0; nt < 4; ++nt) {
      int d = nb + nt * 16 + ln;
      v8bf pk;
#pragma unroll
      for (int v = 0; v < 8; ++v) pk[v] = (__bf16)vacc[nt][v];
      *(v8bf*)(Vt + (size_t)batch * DF * NKEY + (size_t)d * NKEY + key0) = pk;
    }
  }
}

// ---------------------------------------------------------------------------
// Flash attention: one block (8 waves) per 16-query tile; online softmax over
// 2048 keys in chunks of 128.  Waves split keys for QK^T, split d for PV.
// ---------------------------------------------------------------------------
__global__ __launch_bounds__(256)
void flash_attn(const __bf16* __restrict__ Qb, const __bf16* __restrict__ Kb,
                const __bf16* __restrict__ Vt, const int* __restrict__ mask,
                float* __restrict__ out) {
  __shared__ __bf16 Qs[16][DF];       // 16 KB
  __shared__ float  Ss[16][KBLK];     // 8 KB
  __shared__ __bf16 Ps[16][KBLK];     // 4 KB
  __shared__ float  m_s[16], l_s[16], alpha_s[16];

  const int tid  = threadIdx.x;
  const int wave = tid >> 5;
  const int lane = tid & 31;
  const int h    = lane >> 4;
  const int ln   = lane & 15;
  const int b    = blockIdx.x >> 3;
  const int qt   = blockIdx.x & 7;
  const int qrow0 = b * BQ + qt * 16;

  {
    const uint4* src = (const uint4*)(Qb + (size_t)qrow0 * DF);
    uint4* dst = (uint4*)(&Qs[0][0]);
    for (int i = tid; i < 1024; i += 256) dst[i] = src[i];
  }
  if (tid < 16) { m_s[tid] = -3.0e38f; l_s[tid] = 0.0f; }
  __syncthreads();

  v8f oacc[4];
#pragma unroll
  for (int t = 0; t < 4; ++t)
#pragma unroll
    for (int v = 0; v < 8; ++v) oacc[t][v] = 0.0f;

  const float scale = 0.04419417382415922f;  // 1/sqrt(512)
  const __bf16* vbase = Vt + (size_t)b * DF * NKEY;

  for (int kb0 = 0; kb0 < NKEY; kb0 += KBLK) {
    // ---- step 1: this wave's 16x16 score tile ----
    const int keyb = kb0 + wave * 16;
    v8f sacc;
#pragma unroll
    for (int v = 0; v < 8; ++v) sacc[v] = 0.0f;

    const __bf16* krow = Kb + (size_t)(b * NKEY + keyb + ln) * DF;
    for (int fb = 0; fb < DF; fb += 32) {
      v8bf r0 = *(const v8bf*)(&Qs[ln][fb + 16 * h]);
      v8bf r1 = *(const v8bf*)(&Qs[ln][fb + 16 * h + 8]);
      v16bf a = combine8(r0, r1);
      v16bf kk = *(const v16bf*)(krow + fb + 16 * h);
      sacc = wmma_bf16(a, kk, sacc);
    }
    {
      int key = keyb + ln;
      float bias = (1.0f - (float)mask[b * NKEY + key]) * -1.0e10f;
#pragma unroll
      for (int v = 0; v < 8; ++v) {
        int q = v + 8 * h;
        Ss[q][wave * 16 + ln] = sacc[v] * scale + bias;
      }
    }
    __syncthreads();

    // ---- step 2: per-query block max; update running max + alpha ----
    {
      int q = tid >> 4, idx = tid & 15;
      float mx = -3.0e38f;
#pragma unroll
      for (int j = 0; j < 8; ++j) mx = fmaxf(mx, Ss[q][idx * 8 + j]);
#pragma unroll
      for (int off = 1; off < 16; off <<= 1)
        mx = fmaxf(mx, __shfl_xor(mx, off, 32));
      if (idx == 0) {
        float mo = m_s[q];
        float mn = fmaxf(mo, mx);
        m_s[q] = mn;
        alpha_s[q] = __expf(mo - mn);
      }
    }
    __syncthreads();

    // ---- step 3: P = exp(S - m_new) -> bf16 LDS; update running sum ----
    {
      int q = tid >> 4, idx = tid & 15;
      float mn = m_s[q];
      float sum = 0.0f;
#pragma unroll
      for (int j = 0; j < 8; ++j) {
        float p = __expf(Ss[q][idx * 8 + j] - mn);
        Ps[q][idx * 8 + j] = (__bf16)p;
        sum += p;
      }
#pragma unroll
      for (int off = 1; off < 16; off <<= 1) sum += __shfl_xor(sum, off, 32);
      if (idx == 0) l_s[q] = l_s[q] * alpha_s[q] + sum;
    }
    __syncthreads();

    // ---- step 4: rescale O and accumulate P @ V over this wave's d-slice ----
#pragma unroll
    for (int t = 0; t < 4; ++t)
#pragma unroll
      for (int v = 0; v < 8; ++v) oacc[t][v] *= alpha_s[v + 8 * h];

#pragma unroll
    for (int c = 0; c < 4; ++c) {
      v8bf p0 = *(const v8bf*)(&Ps[ln][c * 32 + 16 * h]);
      v8bf p1 = *(const v8bf*)(&Ps[ln][c * 32 + 16 * h + 8]);
      v16bf a = combine8(p0, p1);
#pragma unroll
      for (int nt = 0; nt < 4; ++nt) {
        int d = wave * 64 + nt * 16 + ln;
        v16bf vv = *(const v16bf*)(vbase + (size_t)d * NKEY + kb0 + c * 32 + 16 * h);
        oacc[nt] = wmma_bf16(a, vv, oacc[nt]);
      }
    }
    // next iteration's step-1 barrier serializes LDS reuse.
  }

  // ---- epilogue: O / l, relu, fp32 store ----
#pragma unroll
  for (int nt = 0; nt < 4; ++nt) {
    int d = wave * 64 + nt * 16 + ln;
#pragma unroll
    for (int v = 0; v < 8; ++v) {
      int q = v + 8 * h;
      float val = oacc[nt][v] / l_s[q];
      out[(size_t)(qrow0 + q) * DF + d] = fmaxf(val, 0.0f);
    }
  }
}

extern "C" void kernel_launch(void* const* d_in, const int* in_sizes, int n_in,
                              void* d_out, int out_size, void* d_ws, size_t ws_size,
                              hipStream_t stream) {
  const float* mf   = (const float*)d_in[0];  // [64,128,512]
  const float* x    = (const float*)d_in[1];  // [64,2048,512]
  const int*   mask = (const int*)d_in[2];    // [64,2048]
  const float* Wq   = (const float*)d_in[3];  // [512,512]
  const float* Wk   = (const float*)d_in[4];
  const float* Wv   = (const float*)d_in[5];
  float* out = (float*)d_out;

  const size_t WN = (size_t)DF * DF;          // 262144
  __bf16* Wqb = (__bf16*)d_ws;
  __bf16* Wkb = Wqb + WN;
  __bf16* Wvb = Wkb + WN;
  __bf16* Qb  = Wvb + WN;                     // 8192*512 bf16
  __bf16* Kbf = Qb  + (size_t)8192 * 512;     // 131072*512 bf16
  __bf16* Vt  = Kbf + (size_t)131072 * 512;   // 64*512*2048 bf16

  const int n8 = (int)(WN / 8);               // 32768 -> 128 blocks
  cvt_f32_bf16<<<n8 / 256, 256, 0, stream>>>(Wq, Wqb, n8);
  cvt_f32_bf16<<<n8 / 256, 256, 0, stream>>>(Wk, Wkb, n8);
  cvt_f32_bf16<<<n8 / 256, 256, 0, stream>>>(Wv, Wvb, n8);

  q_gemm<<<8192 / 16, 256, 0, stream>>>(mf, Wqb, Qb);
  kv_gemm<<<131072 / 16, 256, 0, stream>>>(x, Wkb, Wvb, Kbf, Vt);
  flash_attn<<<64 * 8, 256, 0, stream>>>(Qb, Kbf, Vt, mask, out);
}
